// MultiHeadSpectralAttention_55731495633375
// MI455X (gfx1250) — matrix operand order repaired and checked by reading, not compiled, not run on hardware
//
#include <hip/hip_runtime.h>

typedef __attribute__((ext_vector_type(16))) _Float16 v16h;
typedef __attribute__((ext_vector_type(8)))  _Float16 v8h;
typedef __attribute__((ext_vector_type(8)))  float    v8f;

#define D_MODEL 1024
#define NH      16
#define DH      64
#define BATCH   2
#define SEQ     2048
#define MROWS   (BATCH*SEQ)     /* 4096 */
#define QKVC    (3*D_MODEL)     /* 3072 */

static __device__ inline v8f wmma_f16(v16h a, v16h b, v8f c) {
  return __builtin_amdgcn_wmma_f32_16x16x32_f16(false, a, false, b, (short)0, c, false, false);
}

// A fragment (16xK-tile, 16-bit): lane<16 holds row=lane, K in {kb..kb+7, kb+16..kb+23},
// kb = 8*(lane>=16). Two contiguous 8-half (16B) loads per lane.
static __device__ inline v16h load_a_frag(const _Float16* p, int ld, int rm, int k0, int lane) {
  int row = rm + (lane & 15);
  int kb  = k0 + ((lane >> 4) << 3);
  const _Float16* q = p + (size_t)row * ld + kb;
  union { v16h v; v8h h2[2]; } u;
  u.h2[0] = *(const v8h*)(q);
  u.h2[1] = *(const v8h*)(q + 16);
  return u.v;
}

// B fragment (32x16, 16-bit), from K-major storage Bt[n][K]: lane holds N = nb + (lane&15),
// K in {kb..kb+7, kb+16..kb+23}, kb = k0 + 8*(lane>=16). Two contiguous 16B loads per lane.
static __device__ inline v16h load_b_frag(const _Float16* p, int ld, int nb, int k0, int lane) {
  int n  = nb + (lane & 15);
  int kb = k0 + ((lane >> 4) << 3);
  const _Float16* q = p + (size_t)n * ld + kb;
  union { v16h v; v8h h2[2]; } u;
  u.h2[0] = *(const v8h*)(q);
  u.h2[1] = *(const v8h*)(q + 16);
  return u.v;
}

// ---- Prep: circulant spectral-filter kernel c[d], d = (j - i) mod 64 ----
__global__ void k_build_filter(const float* __restrict__ mlog, float* __restrict__ Fc) {
  int d = threadIdx.x;                       // 0..63
  const float w = 6.28318530717958647692f / 64.f;
  float acc = 0.f;
  for (int f = 0; f <= 32; ++f) {
    float m  = 1.f / (1.f + __expf(-mlog[f]));     // sigmoid(mask_logits)
    float wt = (f == 0 || f == 32) ? 1.f : 2.f;
    acc += wt * m * cosf(w * (float)(f * d));
  }
  Fc[d] = acc * (1.f / 64.f);
}

__global__ void k_cvt_x(const float* __restrict__ x, _Float16* __restrict__ xh) {
  int i = blockIdx.x * 256 + threadIdx.x;
  xh[i] = (_Float16)x[i];
}

// w_out [in][out] -> Wot [out][in]  (K-major for B fragments)
__global__ void k_cvt_wout(const float* __restrict__ w, _Float16* __restrict__ wt) {
  int idx = blockIdx.x * 256 + threadIdx.x;  // 1M
  int o = idx >> 10, i = idx & 1023;
  wt[idx] = (_Float16)w[(size_t)i * D_MODEL + o];
}

// Filtered + transposed qkv weight: Wpt[c][r] (c = output col 0..3071, r = in 0..1023).
// q/k columns get the per-head circulant filter folded in; v columns are copied.
__global__ void k_weight_prep(const float* __restrict__ wqkv, const float* __restrict__ Fc,
                              _Float16* __restrict__ wpt) {
  __shared__ float fc[64];
  int tid = threadIdx.x;
  if (tid < 64) fc[tid] = Fc[tid];
  __syncthreads();
  int idx = blockIdx.x * 256 + tid;          // 3072*1024
  int c = idx >> 10, r = idx & 1023;
  float val;
  if (c < 2 * D_MODEL) {
    int j = c & 63;
    int base = c & ~63;
    const float* wrow = wqkv + (size_t)r * QKVC + base;
    float s = 0.f;
    #pragma unroll
    for (int i = 0; i < 64; ++i) s += wrow[i] * fc[(j - i) & 63];
    val = s;
  } else {
    val = wqkv[(size_t)r * QKVC + c];
  }
  wpt[idx] = (_Float16)val;
}

// ---- GEMM1: Qkv[4096x3072] (f16) = Xh[4096x1024] @ Wpt^T, fp32 accum ----
__global__ void k_gemm_qkv(const _Float16* __restrict__ xh, const _Float16* __restrict__ wpt,
                           _Float16* __restrict__ qkv) {
  int w    = blockIdx.x * 8 + (threadIdx.x >> 5);
  int lane = threadIdx.x & 31;
  int mt = (w & 255) * 16;
  int nt = (w >> 8) * 64;
  v8f z = {0.f,0.f,0.f,0.f,0.f,0.f,0.f,0.f};
  v8f acc[4] = {z, z, z, z};
  for (int k0 = 0; k0 < D_MODEL; k0 += 32) {
    v16h a = load_a_frag(xh, D_MODEL, mt, k0, lane);
    #pragma unroll
    for (int f = 0; f < 4; ++f)
      acc[f] = wmma_f16(a, load_b_frag(wpt, D_MODEL, nt + f*16, k0, lane), acc[f]);
  }
  int row0 = mt + ((lane >> 4) << 3);
  int col  = nt + (lane & 15);
  #pragma unroll
  for (int f = 0; f < 4; ++f)
    #pragma unroll
    for (int r = 0; r < 8; ++r)
      qkv[(size_t)(row0 + r) * QKVC + col + f*16] = (_Float16)acc[f][r];
}

// V head-transpose: Vt[b][h][j][s] = Qkv[b*S+s][2048 + h*64 + j]
__global__ void k_transpose_v(const _Float16* __restrict__ qkv, _Float16* __restrict__ vt) {
  int idx = blockIdx.x * 256 + threadIdx.x;  // 2*16*64*2048 = 4M
  int s = idx & (SEQ - 1);
  int j = (idx >> 11) & 63;
  int h = (idx >> 17) & 15;
  int b = idx >> 21;
  vt[idx] = qkv[(size_t)(b * SEQ + s) * QKVC + 2 * D_MODEL + h * DH + j];
}

// ---- Flash attention: one wave per (b, h, 16-query block); 32 keys per iter ----
__global__ void k_attn(const _Float16* __restrict__ qkv, const _Float16* __restrict__ vt,
                       _Float16* __restrict__ o16) {
  __shared__ __align__(16) _Float16 lp[8][16 * 32];
  int wid = threadIdx.x >> 5, lane = threadIdx.x & 31;
  int t  = blockIdx.x * 8 + wid;
  int qt = t & 127;
  int h  = (t >> 7) & 15;
  int b  = t >> 11;
  const _Float16* Qp = qkv + (size_t)(b * SEQ + qt * 16) * QKVC + h * DH;          // filtered Q
  const _Float16* Kp = qkv + (size_t)(b * SEQ) * QKVC + D_MODEL + h * DH;          // filtered K
  const _Float16* Vp = vt + (size_t)((b * NH + h) * DH) * SEQ;                     // V, key-major
  v16h a0 = load_a_frag(Qp, QKVC, 0, 0,  lane);
  v16h a1 = load_a_frag(Qp, QKVC, 0, 32, lane);
  v8f z = {0.f,0.f,0.f,0.f,0.f,0.f,0.f,0.f};
  v8f acc[4] = {z, z, z, z};
  float mrow[8], lrow[8];
  #pragma unroll
  for (int r = 0; r < 8; ++r) { mrow[r] = -3.0e38f; lrow[r] = 0.f; }
  _Float16* myp = &lp[wid][0];

  for (int kk = 0; kk < SEQ; kk += 32) {
    // scores: 16 queries x 32 keys (two 16x16 C tiles, dh=64 -> 2 WMMAs each)
    v8f c0 = z, c1 = z;
    c0 = wmma_f16(a0, load_b_frag(Kp, QKVC, kk,      0,  lane), c0);
    c0 = wmma_f16(a1, load_b_frag(Kp, QKVC, kk,      32, lane), c0);
    c1 = wmma_f16(a0, load_b_frag(Kp, QKVC, kk + 16, 0,  lane), c1);
    c1 = wmma_f16(a1, load_b_frag(Kp, QKVC, kk + 16, 32, lane), c1);
    // streaming softmax; C element r <-> row r+8*(lane>=16); col = lane&15.
    #pragma unroll
    for (int r = 0; r < 8; ++r) {
      float x0 = c0[r] * 0.125f;            // 1/sqrt(64)
      float x1 = c1[r] * 0.125f;
      float tm = fmaxf(x0, x1);
      #pragma unroll
      for (int off = 1; off < 16; off <<= 1) tm = fmaxf(tm, __shfl_xor(tm, off));
      float mn = fmaxf(mrow[r], tm);
      float al = __expf(mrow[r] - mn);
      mrow[r] = mn;
      float p0 = __expf(x0 - mn);
      float p1 = __expf(x1 - mn);
      float rs = p0 + p1;
      #pragma unroll
      for (int off = 1; off < 16; off <<= 1) rs += __shfl_xor(rs, off);
      lrow[r] = lrow[r] * al + rs;
      acc[0][r] *= al; acc[1][r] *= al; acc[2][r] *= al; acc[3][r] *= al;
      int prow = r + ((lane >> 4) << 3);
      myp[prow * 32 + (lane & 15)]      = (_Float16)p0;   // C-layout -> LDS row-major
      myp[prow * 32 + 16 + (lane & 15)] = (_Float16)p1;
    }
    // reload P as an A fragment (16x32) from this wave's private LDS tile
    v16h pa;
    {
      int kb = ((lane >> 4) << 3);
      const _Float16* q = myp + (lane & 15) * 32 + kb;
      union { v16h v; v8h h2[2]; } u;
      u.h2[0] = *(const v8h*)(q);
      u.h2[1] = *(const v8h*)(q + 16);
      pa = u.v;
    }
    #pragma unroll
    for (int f = 0; f < 4; ++f)
      acc[f] = wmma_f16(pa, load_b_frag(Vp, SEQ, f * 16, kk, lane), acc[f]);
  }
  // epilogue: divide by l, write back to [B,S,D] layout
  int row0 = b * SEQ + qt * 16 + ((lane >> 4) << 3);
  int col  = h * DH + (lane & 15);
  #pragma unroll
  for (int r = 0; r < 8; ++r) {
    float inv = 1.f / lrow[r];
    #pragma unroll
    for (int f = 0; f < 4; ++f)
      o16[(size_t)(row0 + r) * D_MODEL + col + f * 16] = (_Float16)(acc[f][r] * inv);
  }
}

// ---- GEMM2: out[4096x1024] (f32) = O16 @ Wot^T + b_out ----
__global__ void k_gemm_out(const _Float16* __restrict__ o16, const _Float16* __restrict__ wot,
                           const float* __restrict__ bias, float* __restrict__ out) {
  int w    = blockIdx.x * 8 + (threadIdx.x >> 5);
  int lane = threadIdx.x & 31;
  int mt = (w & 255) * 16;
  int nt = (w >> 8) * 64;
  v8f z = {0.f,0.f,0.f,0.f,0.f,0.f,0.f,0.f};
  v8f acc[4] = {z, z, z, z};
  for (int k0 = 0; k0 < D_MODEL; k0 += 32) {
    v16h a = load_a_frag(o16, D_MODEL, mt, k0, lane);
    #pragma unroll
    for (int f = 0; f < 4; ++f)
      acc[f] = wmma_f16(a, load_b_frag(wot, D_MODEL, nt + f*16, k0, lane), acc[f]);
  }
  int row0 = mt + ((lane >> 4) << 3);
  int col  = nt + (lane & 15);
  #pragma unroll
  for (int f = 0; f < 4; ++f) {
    float bv = bias[col + f * 16];
    #pragma unroll
    for (int r = 0; r < 8; ++r)
      out[(size_t)(row0 + r) * D_MODEL + col + f * 16] = acc[f][r] + bv;
  }
}

extern "C" void kernel_launch(void* const* d_in, const int* in_sizes, int n_in,
                              void* d_out, int out_size, void* d_ws, size_t ws_size,
                              hipStream_t stream) {
  const float* x    = (const float*)d_in[0];
  const float* wqkv = (const float*)d_in[1];
  const float* wout = (const float*)d_in[2];
  const float* bout = (const float*)d_in[3];
  const float* mlog = (const float*)d_in[4];

  char* ws = (char*)d_ws;
  size_t off = 0;
  float*    Fc  = (float*)(ws + off);    off += 256;
  _Float16* Xh  = (_Float16*)(ws + off); off += (size_t)MROWS * D_MODEL * 2;          // 8 MB
  _Float16* Wpt = (_Float16*)(ws + off); off += (size_t)QKVC * D_MODEL * 2;           // 6 MB
  _Float16* Wot = (_Float16*)(ws + off); off += (size_t)D_MODEL * D_MODEL * 2;        // 2 MB
  _Float16* Qkv = (_Float16*)(ws + off); off += (size_t)MROWS * QKVC * 2;             // 24 MB
  _Float16* Vt  = (_Float16*)(ws + off); off += (size_t)BATCH * NH * DH * SEQ * 2;    // 8 MB
  _Float16* O16 = (_Float16*)(ws + off); off += (size_t)MROWS * D_MODEL * 2;          // 8 MB

  k_build_filter<<<1, 64, 0, stream>>>(mlog, Fc);
  k_cvt_x<<<(MROWS * D_MODEL) / 256, 256, 0, stream>>>(x, Xh);
  k_cvt_wout<<<(D_MODEL * D_MODEL) / 256, 256, 0, stream>>>(wout, Wot);
  k_weight_prep<<<(QKVC * D_MODEL) / 256, 256, 0, stream>>>(wqkv, Fc, Wpt);
  k_gemm_qkv<<<(256 * 48) / 8, 256, 0, stream>>>(Xh, Wpt, Qkv);
  k_transpose_v<<<(BATCH * NH * DH * SEQ) / 256, 256, 0, stream>>>(Qkv, Vt);
  k_attn<<<(BATCH * NH * (SEQ / 16)) / 8, 256, 0, stream>>>(Qkv, Vt, O16);
  k_gemm_out<<<(256 * 16) / 8, 256, 0, stream>>>(O16, Wot, bout, (float*)d_out);
}
